// Agent_21577915695771
// MI455X (gfx1250) — compile-verified, matrix operand order checked by reading, not run on hardware
//
#include <hip/hip_runtime.h>
#include <math.h>

typedef _Float16 h16;
typedef __attribute__((ext_vector_type(16))) _Float16 v16h;
typedef __attribute__((ext_vector_type(8)))  _Float16 v8h;
typedef __attribute__((ext_vector_type(8)))  float    v8f;

// ---------------------------------------------------------------------------
// WMMA helpers (CDNA5 wave32, 16x16x32 f16 -> f32)
// ---------------------------------------------------------------------------
static __device__ __forceinline__ v8f wmma16(v16h a, v16h b, v8f c) {
  return __builtin_amdgcn_wmma_f32_16x16x32_f16(false, a, false, b, (short)0, c,
                                                false, false);
}

// A fragment: 16x32 f16, row-major source with leading dim lda.
// lanes 0-15: row=lane, K in {0..7,16..23}; lanes 16-31: row=lane-16,
// K in {8..15,24..31}. Contiguous 8-halves runs -> ds_load_b128.
static __device__ __forceinline__ v16h frag_a(const h16* A, int lda) {
  const int lane = threadIdx.x & 31;
  const int r    = lane & 15;
  const int ko   = (lane & 16) ? 8 : 0;
  const h16* p = A + r * lda + ko;
  v16h f;
#pragma unroll
  for (int j = 0; j < 8; ++j) f[j] = p[j];
#pragma unroll
  for (int j = 0; j < 8; ++j) f[8 + j] = p[16 + j];
  return f;
}

// B matrices are pre-swizzled fragment-major: tile (kt,nt) of 32x16 is stored
// as 32 lanes x 16 contiguous f16 (32B per lane). One v16h load per fragment.
// Swizzle rule per lane/j: ko = (lane&16)?8:0 ; kr = (j<8) ? ko+j : ko+j+8 ;
// n = nt*16 + (lane&15).

// ---------------------------------------------------------------------------
// A-operand gather functors (implicit im2col; gather+f32->f16 into LDS)
// ---------------------------------------------------------------------------
struct AGPlain {                       // row-major f16 matrix [M, ld]
  const h16* p; int ld;
  __device__ float operator()(int r, int k) const {
    return (float)p[(size_t)r * ld + k];
  }
};
struct AGFc {                          // NCHW-flat view of NHWC act3 [4096,7,7,64]
  const h16* p;
  __device__ float operator()(int r, int k) const {
    int c = k / 49, t = k - c * 49;
    return (float)p[(((size_t)r * 7 + t / 7) * 7 + (t % 7)) * 64 + c];
  }
};
struct AGConvX {                       // conv1 from raw frames (f32 NCHW, IC=1, k=8, s=4)
  const float* x;
  __device__ float operator()(int row, int k) const {
    int img = row / 400, t = row - img * 400;
    int oy = t / 20, ox = t - oy * 20;
    int ky = k >> 3, kx = k & 7;
    return x[((size_t)img * 84 + oy * 4 + ky) * 84 + (ox * 4 + kx)] * (1.0f / 255.0f);
  }
};
struct AGConv {                        // VALID conv over NHWC f16 input
  const h16* p; int IC, IH, IW, KW, S, OH, OW;
  __device__ float operator()(int row, int k) const {
    int hw = OH * OW;
    int img = row / hw, t = row - img * hw;
    int oy = t / OW, ox = t - oy * OW;
    int ic = k % IC, q = k / IC;
    int kx = q % KW, ky = q / KW;
    return (float)p[(((size_t)img * IH + (oy * S + ky)) * IW + (ox * S + kx)) * IC + ic];
  }
};
struct AGDeconvH {                     // ConvTranspose over NHWC f16 (lhs-dilated conv, pad K-1)
  const h16* p; int IC, IH, IW, KH, KW, S, OH, OW;
  __device__ float operator()(int row, int k) const {
    int hw = OH * OW;
    int img = row / hw, t = row - img * hw;
    int oy = t / OW, ox = t - oy * OW;
    int ic = k % IC, q = k / IC;
    int kx = q % KW, ky = q / KW;
    int u = oy + ky - (KH - 1), v = ox + kx - (KW - 1);
    if (u < 0 || v < 0) return 0.0f;
    if ((u % S) || (v % S)) return 0.0f;
    int iy = u / S, ix = v / S;
    if (iy >= IH || ix >= IW) return 0.0f;
    return (float)p[(((size_t)img * IH + iy) * IW + ix) * IC + ic];
  }
};
struct AGDeconvN {                     // ConvTranspose over NCHW-flat f16 [img][IC*IH*IW]
  const h16* p; int IC, IH, IW, KH, KW, S, OH, OW;
  __device__ float operator()(int row, int k) const {
    int hw = OH * OW;
    int img = row / hw, t = row - img * hw;
    int oy = t / OW, ox = t - oy * OW;
    int ic = k % IC, q = k / IC;
    int kx = q % KW, ky = q / KW;
    int u = oy + ky - (KH - 1), v = ox + kx - (KW - 1);
    if (u < 0 || v < 0) return 0.0f;
    if ((u % S) || (v % S)) return 0.0f;
    int iy = u / S, ix = v / S;
    if (iy >= IH || ix >= IW) return 0.0f;
    return (float)p[(size_t)img * (IC * IH * IW) + ((ic * IH + iy) * IW + ix)];
  }
};

// ---------------------------------------------------------------------------
// C writers
// ---------------------------------------------------------------------------
struct CWh16 {
  h16* o; int ld;
  __device__ void operator()(int r, int c, float v) const { o[(size_t)r * ld + c] = (h16)v; }
};
struct CWf32 {
  float* o; int ld;
  __device__ void operator()(int r, int c, float v) const { o[(size_t)r * ld + c] = v; }
};
struct CWrecon {                       // N==1: only col 0 reaches here
  float* o;
  __device__ void operator()(int r, int c, float v) const { o[r] = v; }
};

// ---------------------------------------------------------------------------
// Generic WMMA GEMM core: block = 256 threads = 8 waves.
// Block owns one 16-row M tile (staged to LDS once per K-chunk, b128 stores);
// wave w owns N tile (blockIdx.y*8 + w). B is fragment-major swizzled.
// All layer Ms are multiples of 16 and grids cover M exactly, so there are
// no row guards anywhere -> straight-line epilogue (8 coalesced stores).
// ---------------------------------------------------------------------------
template <bool RELU, typename AG, typename CW>
static __device__ void gemm_core(const AG& ag, const CW& cw,
                                 const h16* __restrict__ B,
                                 const float* __restrict__ bias,
                                 int M, int N, int Npad, int K, int KC, h16* lds) {
  const int tid    = threadIdx.x;
  const int lane   = tid & 31;
  const int wave   = tid >> 5;
  const int mrow0  = blockIdx.x * 16;
  const int ntiles = Npad >> 4;
  const int ntile  = blockIdx.y * 8 + wave;
  const bool active = ntile < ntiles;
  v8f acc = {};
  for (int k0 = 0; k0 < K; k0 += KC) {
    const int kc = (KC < (K - k0)) ? KC : (K - k0);
    // stage A[16 x KC] into LDS, 8 f16 per thread per iteration (one b128 store)
    for (int i0 = tid * 8; i0 < 16 * KC; i0 += 2048) {
      int r = i0 / KC, k = i0 - r * KC;
      int row = mrow0 + r;
      v8h pack;
#pragma unroll
      for (int j = 0; j < 8; ++j) {
        int kk = k + j;
        pack[j] = (h16)((kk < kc) ? ag(row, k0 + kk) : 0.0f);
      }
      *(v8h*)(lds + r * KC + k) = pack;
    }
    __syncthreads();
    if (active) {
      const h16* Bp = B + (((size_t)(k0 >> 5) * ntiles + ntile) << 9) + lane * 16;
      const size_t bstep = (size_t)ntiles << 9;
      for (int kk = 0; kk < kc; kk += 32) {
        if (kk + 32 < kc) __builtin_prefetch(Bp + bstep, 0, 1);
        v16h a = frag_a(lds + kk, KC);
        v16h b = *(const v16h*)Bp;
        acc = wmma16(a, b, acc);
        Bp += bstep;
      }
    }
    __syncthreads();
  }
  if (active) {
    const int cl  = lane & 15;
    const int r8  = (lane & 16) ? 8 : 0;
    const int col = ntile * 16 + cl;
    if (col < N) {
      const float bv = bias[col];
#pragma unroll
      for (int r = 0; r < 8; ++r) {
        float v = acc[r] + bv;
        if (RELU) v = fmaxf(v, 0.0f);
        cw(mrow0 + r + r8, col, v);
      }
    }
  }
}

// ---------------------------------------------------------------------------
// Per-layer GEMM kernels (all dims are compile-time constants of the model)
// ---------------------------------------------------------------------------
__global__ void __launch_bounds__(256) k_conv1(const float* x, const h16* B, const float* bias, h16* out) {
  extern __shared__ h16 lds[];
  gemm_core<true>(AGConvX{x}, CWh16{out, 32}, B, bias, 1638400, 32, 32, 64, 64, lds);
}
__global__ void __launch_bounds__(256) k_conv2(const h16* a, const h16* B, const float* bias, h16* out) {
  extern __shared__ h16 lds[];
  gemm_core<true>(AGConv{a, 32, 20, 20, 4, 2, 9, 9}, CWh16{out, 64}, B, bias, 331776, 64, 64, 512, 512, lds);
}
__global__ void __launch_bounds__(256) k_conv3(const h16* a, const h16* B, const float* bias, h16* out) {
  extern __shared__ h16 lds[];
  gemm_core<true>(AGConv{a, 64, 9, 9, 3, 1, 7, 7}, CWh16{out, 64}, B, bias, 200704, 64, 64, 576, 576, lds);
}
__global__ void __launch_bounds__(256) k_encfc(const h16* a, const h16* B, const float* bias, h16* out) {
  extern __shared__ h16 lds[];
  gemm_core<true>(AGFc{a}, CWh16{out, 256}, B, bias, 4096, 256, 256, 3136, 1024, lds);
}
__global__ void __launch_bounds__(256) k_xg(const h16* a, const h16* B, const float* bias, float* out) {
  extern __shared__ h16 lds[];
  gemm_core<false>(AGPlain{a, 256}, CWf32{out, 2048}, B, bias, 4096, 2048, 2048, 256, 256, lds);
}
__global__ void __launch_bounds__(256) k_decfc(const h16* a, const h16* B, const float* bias, h16* out) {
  extern __shared__ h16 lds[];
  gemm_core<true>(AGPlain{a, 512}, CWh16{out, 256}, B, bias, 4096, 256, 256, 512, 512, lds);
}
__global__ void __launch_bounds__(256) k_declin(const h16* a, const h16* B, const float* bias, h16* out) {
  extern __shared__ h16 lds[];
  gemm_core<true>(AGPlain{a, 256}, CWh16{out, 3136}, B, bias, 4096, 3136, 3136, 256, 256, lds);
}
__global__ void __launch_bounds__(256) k_dct1(const h16* a, const h16* B, const float* bias, h16* out) {
  extern __shared__ h16 lds[];
  gemm_core<true>(AGDeconvN{a, 64, 7, 7, 3, 3, 1, 9, 9}, CWh16{out, 64}, B, bias, 331776, 64, 64, 576, 576, lds);
}
__global__ void __launch_bounds__(256) k_dct2(const h16* a, const h16* B, const float* bias, h16* out) {
  extern __shared__ h16 lds[];
  gemm_core<true>(AGDeconvH{a, 64, 9, 9, 4, 4, 2, 20, 20}, CWh16{out, 32}, B, bias, 1638400, 32, 32, 1024, 1024, lds);
}
__global__ void __launch_bounds__(256) k_dct3(const h16* a, const h16* B, const float* bias, float* out) {
  extern __shared__ h16 lds[];
  gemm_core<true>(AGDeconvH{a, 32, 20, 20, 8, 8, 4, 84, 84}, CWrecon{out}, B, bias, 28901376, 1, 16, 2048, 1024, lds);
}

// ---------------------------------------------------------------------------
// Persistent masked-LSTM kernel: 1 block, 32 waves. h (f16) lives in LDS as the
// WMMA A operand; c lives in accumulator-layout registers. 128 serial steps.
// W_hh is fragment-major swizzled [16 ktiles x 128 ntiles x 512].
// ---------------------------------------------------------------------------
__global__ void __launch_bounds__(1024) k_lstm(
    const float* __restrict__ xg, const h16* __restrict__ Bhh,
    const float* __restrict__ h0, const float* __restrict__ c0,
    const float* __restrict__ done, h16* __restrict__ hid,
    float* __restrict__ hN, float* __restrict__ cN) {
  __shared__ h16 hbuf[16384];   // 32 x 512 f16
  __shared__ float dm[32];
  const int tid  = threadIdx.x;
  const int lane = tid & 31;
  const int w    = tid >> 5;
  const int mt   = w >> 4;          // row tile: rows [mt*16, mt*16+16)
  const int jt0  = (w & 15) * 2;    // this wave's two hidden 16-col tiles
  const int r8   = (lane & 16) ? 8 : 0;
  const int cl   = lane & 15;

  float c[2][8];
  for (int i = tid; i < 16384; i += 1024) hbuf[i] = (h16)h0[i];
#pragma unroll
  for (int pi = 0; pi < 2; ++pi)
#pragma unroll
    for (int r = 0; r < 8; ++r) {
      int row = mt * 16 + r + r8;
      c[pi][r] = c0[row * 512 + (jt0 + pi) * 16 + cl];
    }
  __syncthreads();

  for (int t = 0; t < 128; ++t) {
    if (tid < 32) dm[tid] = 1.0f - done[t * 32 + tid];
    __syncthreads();
    for (int i = tid; i < 16384; i += 1024)
      hbuf[i] = (h16)((float)hbuf[i] * dm[i >> 9]);
#pragma unroll
    for (int pi = 0; pi < 2; ++pi)
#pragma unroll
      for (int r = 0; r < 8; ++r) c[pi][r] *= dm[mt * 16 + r + r8];
    __syncthreads();

    v8f zero = {};
    v8f acc[2][4];
#pragma unroll
    for (int pi = 0; pi < 2; ++pi)
#pragma unroll
      for (int g = 0; g < 4; ++g) acc[pi][g] = zero;

    for (int kk = 0; kk < 512; kk += 32) {
      v16h a = frag_a(hbuf + mt * 16 * 512 + kk, 512);
      const size_t ktbase = (size_t)(kk >> 5) * 128;
#pragma unroll
      for (int pi = 0; pi < 2; ++pi)
#pragma unroll
        for (int g = 0; g < 4; ++g) {
          const h16* bp = Bhh + ((ktbase + (g * 32 + jt0 + pi)) << 9) + lane * 16;
          v16h b = *(const v16h*)bp;
          acc[pi][g] = wmma16(a, b, acc[pi][g]);
        }
    }
    __syncthreads();   // all reads of h done before rewriting it

#pragma unroll
    for (int pi = 0; pi < 2; ++pi) {
      const int col = (jt0 + pi) * 16 + cl;
#pragma unroll
      for (int r = 0; r < 8; ++r) {
        const int row = mt * 16 + r + r8;
        const float* xr = xg + ((size_t)t * 32 + row) * 2048;
        float gi = 1.0f / (1.0f + expf(-(acc[pi][0][r] + xr[col])));
        float gf = 1.0f / (1.0f + expf(-(acc[pi][1][r] + xr[512 + col])));
        float gg = tanhf(acc[pi][2][r] + xr[1024 + col]);
        float go = 1.0f / (1.0f + expf(-(acc[pi][3][r] + xr[1536 + col])));
        float cn = gf * c[pi][r] + gi * gg;
        c[pi][r] = cn;
        float hh = go * tanhf(cn);
        hbuf[row * 512 + col] = (h16)hh;
        hid[((size_t)t * 32 + row) * 512 + col] = (h16)hh;
        if (t == 127) {
          hN[row * 512 + col] = hh;
          cN[row * 512 + col] = cn;
        }
      }
    }
    __syncthreads();
  }
}

// ---------------------------------------------------------------------------
// Heads: one wave per row; 18 logits + value via shuffle reductions,
// then log-softmax / entropy / gather.
// ---------------------------------------------------------------------------
__global__ void __launch_bounds__(32) k_heads(
    const h16* __restrict__ hid, const float* __restrict__ aw,
    const float* __restrict__ ab, const float* __restrict__ cwt,
    const float* __restrict__ cb, const int* __restrict__ action,
    float* __restrict__ lp, float* __restrict__ ent, float* __restrict__ val) {
  const int row  = blockIdx.x;
  const int lane = threadIdx.x;
  float h[16];
#pragma unroll
  for (int j = 0; j < 16; ++j) h[j] = (float)hid[(size_t)row * 512 + lane + j * 32];
  float logits[18];
  for (int o = 0; o < 18; ++o) {
    float s = 0.0f;
#pragma unroll
    for (int j = 0; j < 16; ++j) s += h[j] * aw[o * 512 + lane + j * 32];
    for (int off = 16; off > 0; off >>= 1) s += __shfl_xor(s, off, 32);
    logits[o] = s + ab[o];
  }
  float v = 0.0f;
#pragma unroll
  for (int j = 0; j < 16; ++j) v += h[j] * cwt[lane + j * 32];
  for (int off = 16; off > 0; off >>= 1) v += __shfl_xor(v, off, 32);
  v += cb[0];
  float mx = logits[0];
  for (int o = 1; o < 18; ++o) mx = fmaxf(mx, logits[o]);
  float se = 0.0f;
  for (int o = 0; o < 18; ++o) se += expf(logits[o] - mx);
  float lz = mx + logf(se);
  float e = 0.0f;
  for (int o = 0; o < 18; ++o) { float l = logits[o] - lz; e -= expf(l) * l; }
  if (lane == 0) {
    val[row] = v;
    ent[row] = e;
    int a = action[row];
    float sel = logits[0];
    for (int o = 1; o < 18; ++o) if (o == a) sel = logits[o];
    lp[row] = sel - lz;
  }
}

// ---------------------------------------------------------------------------
// Weight conversion: build f16 B matrices in FRAGMENT-MAJOR swizzled layout.
// dst index i decomposes as [tile = kt*ntiles + nt][lane 0..31][j 0..15];
// (lane, j) -> (kr, n%16) per the WMMA B VGPR layout, so each wave lane loads
// its whole fragment as one contiguous 32-byte v16h.
// ---------------------------------------------------------------------------
__global__ void k_bfc(h16* dst, const float* w, int K, int N, int Npad) {
  int i = blockIdx.x * 256 + threadIdx.x;
  if (i >= K * Npad) return;
  int j = i & 15, lane = (i >> 4) & 31, tile = i >> 9;
  int ntiles = Npad >> 4;
  int nt = tile % ntiles, kt = tile / ntiles;
  int ko = (lane & 16) ? 8 : 0;
  int kr = (j < 8) ? (ko + j) : (ko + j + 8);
  int k = kt * 32 + kr;
  int n = nt * 16 + (lane & 15);
  dst[i] = (h16)((n < N && k < K) ? w[(size_t)n * K + k] : 0.0f);
}
__global__ void k_bconv(h16* dst, const float* w, int O, int I, int KH, int KW,
                        int Npad, int flip) {
  int K = KH * KW * I;
  int i = blockIdx.x * 256 + threadIdx.x;
  if (i >= K * Npad) return;
  int j = i & 15, lane = (i >> 4) & 31, tile = i >> 9;
  int ntiles = Npad >> 4;
  int nt = tile % ntiles, kt = tile / ntiles;
  int ko = (lane & 16) ? 8 : 0;
  int kr = (j < 8) ? (ko + j) : (ko + j + 8);
  int k = kt * 32 + kr;
  int n = nt * 16 + (lane & 15);
  int ic = k % I, q = k / I;
  int kx = q % KW, ky = q / KW;
  float v = 0.0f;
  if (n < O && k < K) {
    if (!flip) v = w[(((size_t)n * I + ic) * KH + ky) * KW + kx];
    else       v = w[(((size_t)ic * O + n) * KH + (KH - 1 - ky)) * KW + (KW - 1 - kx)];
  }
  dst[i] = (h16)v;
}
__global__ void k_bias2(float* dst, const float* a, const float* b) {
  int i = blockIdx.x * 256 + threadIdx.x;
  if (i < 2048) dst[i] = a[i] + b[i];
}

// ---------------------------------------------------------------------------
// Host orchestration
// ---------------------------------------------------------------------------
extern "C" void kernel_launch(void* const* d_in, const int* in_sizes, int n_in,
                              void* d_out, int out_size, void* d_ws, size_t ws_size,
                              hipStream_t stream) {
  (void)in_sizes; (void)n_in; (void)out_size; (void)ws_size;
  const float* x        = (const float*)d_in[0];
  const float* done     = (const float*)d_in[1];
  const int*   action   = (const int*)d_in[2];
  const float* h0       = (const float*)d_in[3];
  const float* c0       = (const float*)d_in[4];
  const float* conv1_w  = (const float*)d_in[5];
  const float* conv1_b  = (const float*)d_in[6];
  const float* conv2_w  = (const float*)d_in[7];
  const float* conv2_b  = (const float*)d_in[8];
  const float* conv3_w  = (const float*)d_in[9];
  const float* conv3_b  = (const float*)d_in[10];
  const float* enc_fc_w = (const float*)d_in[11];
  const float* enc_fc_b = (const float*)d_in[12];
  const float* w_ih     = (const float*)d_in[13];
  const float* w_hh     = (const float*)d_in[14];
  const float* b_ih     = (const float*)d_in[15];
  const float* b_hh     = (const float*)d_in[16];
  const float* actor_w  = (const float*)d_in[17];
  const float* actor_b  = (const float*)d_in[18];
  const float* critic_w = (const float*)d_in[19];
  const float* critic_b = (const float*)d_in[20];
  const float* dec_fc_w = (const float*)d_in[21];
  const float* dec_fc_b = (const float*)d_in[22];
  const float* dec_lin_w= (const float*)d_in[23];
  const float* dec_lin_b= (const float*)d_in[24];
  const float* dct1_w   = (const float*)d_in[25];
  const float* dct1_b   = (const float*)d_in[26];
  const float* dct2_w   = (const float*)d_in[27];
  const float* dct2_b   = (const float*)d_in[28];
  const float* dct3_w   = (const float*)d_in[29];
  const float* dct3_b   = (const float*)d_in[30];

  // outputs: [log_prob(4096) | entropy(4096) | value(4096) | recon(4096*7056) | hN(16384) | cN(16384)]
  float* out   = (float*)d_out;
  float* o_lp  = out;
  float* o_ent = out + 4096;
  float* o_val = out + 8192;
  float* o_rec = out + 12288;
  float* o_hn  = out + 12288 + (size_t)4096 * 7056;
  float* o_cn  = o_hn + 16384;

  // workspace layout (dead live-ranges aliased: actA=act1/dct2out, actB=act2/dct1out,
  // actC=act3/d2, actZ=emb/d1)
  char* wsb = (char*)d_ws;
  size_t off = 0;
  auto alloc = [&](size_t bytes) -> void* {
    void* p = wsb + off;
    off += (bytes + 255) & ~(size_t)255;
    return p;
  };
  h16* Bc1 = (h16*)alloc((size_t)64 * 32 * 2);
  h16* Bc2 = (h16*)alloc((size_t)512 * 64 * 2);
  h16* Bc3 = (h16*)alloc((size_t)576 * 64 * 2);
  h16* Bfc = (h16*)alloc((size_t)3136 * 256 * 2);
  h16* Bih = (h16*)alloc((size_t)256 * 2048 * 2);
  h16* Bhh = (h16*)alloc((size_t)512 * 2048 * 2);
  h16* Bdf = (h16*)alloc((size_t)512 * 256 * 2);
  h16* Bdl = (h16*)alloc((size_t)256 * 3136 * 2);
  h16* Bd1 = (h16*)alloc((size_t)576 * 64 * 2);
  h16* Bd2 = (h16*)alloc((size_t)1024 * 32 * 2);
  h16* Bd3 = (h16*)alloc((size_t)2048 * 16 * 2);
  float* bsum = (float*)alloc((size_t)2048 * 4);
  h16* actA = (h16*)alloc((size_t)1638400 * 32 * 2);
  h16* actB = (h16*)alloc((size_t)331776 * 64 * 2);
  h16* actC = (h16*)alloc((size_t)4096 * 3136 * 2);
  h16* actZ = (h16*)alloc((size_t)4096 * 256 * 2);
  float* xg = (float*)alloc((size_t)4096 * 2048 * 4);
  h16* hid  = (h16*)alloc((size_t)4096 * 512 * 2);

  auto cdiv = [](long long a, long long b) { return (unsigned)((a + b - 1) / b); };

  // ---- weight conversion (to fragment-major swizzled f16 B) ----
  k_bconv<<<cdiv(8 * 8 * 1 * 32, 256), 256, 0, stream>>>(Bc1, conv1_w, 32, 1, 8, 8, 32, 0);
  k_bconv<<<cdiv(4 * 4 * 32 * 64, 256), 256, 0, stream>>>(Bc2, conv2_w, 64, 32, 4, 4, 64, 0);
  k_bconv<<<cdiv(3 * 3 * 64 * 64, 256), 256, 0, stream>>>(Bc3, conv3_w, 64, 64, 3, 3, 64, 0);
  k_bfc<<<cdiv((size_t)3136 * 256, 256), 256, 0, stream>>>(Bfc, enc_fc_w, 3136, 256, 256);
  k_bfc<<<cdiv((size_t)256 * 2048, 256), 256, 0, stream>>>(Bih, w_ih, 256, 2048, 2048);
  k_bfc<<<cdiv((size_t)512 * 2048, 256), 256, 0, stream>>>(Bhh, w_hh, 512, 2048, 2048);
  k_bfc<<<cdiv((size_t)512 * 256, 256), 256, 0, stream>>>(Bdf, dec_fc_w, 512, 256, 256);
  k_bfc<<<cdiv((size_t)256 * 3136, 256), 256, 0, stream>>>(Bdl, dec_lin_w, 256, 3136, 3136);
  k_bconv<<<cdiv(3 * 3 * 64 * 64, 256), 256, 0, stream>>>(Bd1, dct1_w, 64, 64, 3, 3, 64, 1);
  k_bconv<<<cdiv(4 * 4 * 64 * 32, 256), 256, 0, stream>>>(Bd2, dct2_w, 32, 64, 4, 4, 32, 1);
  k_bconv<<<cdiv(8 * 8 * 32 * 16, 256), 256, 0, stream>>>(Bd3, dct3_w, 1, 32, 8, 8, 16, 1);
  k_bias2<<<8, 256, 0, stream>>>(bsum, b_ih, b_hh);

  // ---- encoder ----
  k_conv1<<<dim3(102400, 1), 256, 16 * 64 * 2, stream>>>(x, Bc1, conv1_b, actA);
  k_conv2<<<dim3(20736, 1), 256, 16 * 512 * 2, stream>>>(actA, Bc2, conv2_b, actB);
  k_conv3<<<dim3(12544, 1), 256, 16 * 576 * 2, stream>>>(actB, Bc3, conv3_b, actC);
  k_encfc<<<dim3(256, 2), 256, 16 * 1024 * 2, stream>>>(actC, Bfc, enc_fc_b, actZ);

  // ---- LSTM (x-projection batched, then serial recurrence) ----
  k_xg<<<dim3(256, 16), 256, 16 * 256 * 2, stream>>>(actZ, Bih, bsum, xg);
  k_lstm<<<1, 1024, 0, stream>>>(xg, Bhh, h0, c0, done, hid, o_hn, o_cn);

  // ---- heads ----
  k_heads<<<4096, 32, 0, stream>>>(hid, actor_w, actor_b, critic_w, critic_b,
                                   action, o_lp, o_ent, o_val);

  // ---- decoder ----
  k_decfc<<<dim3(256, 2), 256, 16 * 512 * 2, stream>>>(hid, Bdf, dec_fc_b, actZ);
  k_declin<<<dim3(256, 25), 256, 16 * 256 * 2, stream>>>(actZ, Bdl, dec_lin_b, actC);
  k_dct1<<<dim3(20736, 1), 256, 16 * 576 * 2, stream>>>(actC, Bd1, dct1_b, actB);
  k_dct2<<<dim3(102400, 1), 256, 16 * 1024 * 2, stream>>>(actB, Bd2, dct2_b, actA);
  k_dct3<<<dim3(1806336, 1), 256, 16 * 1024 * 2, stream>>>(actA, Bd3, dct3_b, o_rec);
}
